// MultiHeadAttention_54339926229039
// MI455X (gfx1250) — matrix operand order repaired and checked by reading, not compiled, not run on hardware
//
#include <hip/hip_runtime.h>

#define DEVINL __device__ __forceinline__

typedef __attribute__((ext_vector_type(16))) __bf16 v16bf;
typedef __attribute__((ext_vector_type(8)))  __bf16 v8bf;
typedef __attribute__((ext_vector_type(8)))  float  v8f;

constexpr int B_   = 4;
constexpr int S_   = 2048;
constexpr int D_   = 1024;
constexpr int H_   = 16;
constexpr int DK_  = 64;
constexpr int MTOT = B_ * S_;   // 8192 rows

// ---------------------------------------------------------------------------
// CDNA5 async global->LDS copy (ASYNCcnt path), guarded by __has_builtin.
// Builtin signature (from hipcc diagnostic): param0 = AS1 pointer to
// `int __attribute__((vector_size(16)))`, param1 = LDS-side pointer, then
// imm offset + cpol.
// ---------------------------------------------------------------------------
#if defined(__has_builtin)
#  if __has_builtin(__builtin_amdgcn_global_load_async_to_lds_b128) && \
      __has_builtin(__builtin_amdgcn_s_wait_asynccnt)
#    define USE_ASYNC_LDS 1
#  endif
#endif
#ifndef USE_ASYNC_LDS
#  define USE_ASYNC_LDS 0
#endif

#if USE_ASYNC_LDS
typedef int v4i __attribute__((vector_size(16)));
typedef __attribute__((address_space(1))) v4i* gas_v4i_ptr;
typedef __attribute__((address_space(3))) v4i* las_v4i_ptr;

DEVINL void async_cp16B(const __bf16* g, __bf16* l) {
  __builtin_amdgcn_global_load_async_to_lds_b128(
      (gas_v4i_ptr)(uintptr_t)g,
      (las_v4i_ptr)(void*)l,
      0, 0);
}
DEVINL void wait_async0() { __builtin_amdgcn_s_wait_asynccnt(0); }
#else
DEVINL void wait_async0() {}
#endif

// ---------------------------------------------------------------------------
// WMMA: D = A(16x32 bf16) * B(32x16 bf16) + C(16x16 f32)
// ---------------------------------------------------------------------------
DEVINL v8f wmma_bf16(v16bf a, v16bf b, v8f c) {
  return __builtin_amdgcn_wmma_f32_16x16x32_bf16(
      false, a, false, b, (short)0, c, false, false);
}

// A fragment (16x32 bf16) from row-major tile p[m*ld + k]; 2x ds_load_b128.
DEVINL v16bf load_A(const __bf16* p, int ld) {
  const int lane = threadIdx.x & 31;
  const int m  = lane & 15;
  const int kb = (lane >> 4) << 3;
  v16bf a;
#pragma unroll
  for (int h = 0; h < 16; ++h) {
    const int k = ((h >> 3) << 4) + kb + (h & 7);
    a[h] = p[m * ld + k];
  }
  return a;
}

// B fragment (32x16 bf16) from transposed storage T[n][k]==B[k][n]; 2x b128.
DEVINL v16bf load_BT(const __bf16* p, int ld) {
  const int lane = threadIdx.x & 31;
  const int n  = lane & 15;
  const int kb = (lane >> 4) << 3;
  v16bf b;
#pragma unroll
  for (int h = 0; h < 16; ++h) b[h] = p[n * ld + ((h >> 3) << 4) + kb + (h & 7)];
  return b;
}

// ---------------------------------------------------------------------------
// GEMM: Y = X @ W^T + bias.  X:[MTOT,1024] (fp32 or bf16), W:[1024,1024] fp32.
// MODE 0: bf16 -> head-major [B,H,S,DK]       (Q,K projections)
// MODE 1: fp32 -> [MTOT,1024]                 (output projection)
// MODE 2: bf16 -> transposed head [B,H,DK,S]  (V projection)
// Block tile 128x128, BK=64, double-buffered LDS + register-staged global
// loads for tile k+1 issued before the WMMAs on tile k (1 barrier / k-step).
// ---------------------------------------------------------------------------
template <typename TX, int MODE>
__global__ __launch_bounds__(256)
void gemm_xwt(const TX* __restrict__ X, const float* __restrict__ W,
              const float* __restrict__ bias, void* __restrict__ Y) {
  constexpr int BM = 128, BN = 128, BK = 64, LDT = BK + 8, NIT = D_ / BK;
  __shared__ __bf16 As[2][BM][LDT];
  __shared__ __bf16 Ws[2][BN][LDT];

  const int tid  = threadIdx.x;
  const int wave = tid >> 5;
  const int wm   = wave & 3;
  const int wn   = wave >> 2;
  const int m0   = blockIdx.y * BM;
  const int n0   = blockIdx.x * BN;

  v8f acc[2][4];
#pragma unroll
  for (int i = 0; i < 2; ++i)
#pragma unroll
    for (int j = 0; j < 4; ++j) acc[i][j] = {};

  const int lr = tid >> 1;                   // 0..127
  const int lc = (tid & 1) * 32;             // 0 or 32
  const TX*    xrow = X + (long)(m0 + lr) * D_ + lc;
  const float* wrow = W + (long)(n0 + lr) * D_ + lc;

  float xr[32], wr[32];
  auto gload = [&](int k0) {
#pragma unroll
    for (int j = 0; j < 32; ++j) { xr[j] = (float)xrow[k0 + j]; wr[j] = wrow[k0 + j]; }
  };
  auto lstore = [&](int b) {
#pragma unroll
    for (int j = 0; j < 32; ++j) {
      As[b][lr][lc + j] = (__bf16)xr[j];
      Ws[b][lr][lc + j] = (__bf16)wr[j];
    }
  };

  gload(0);
  lstore(0);
  __syncthreads();

  for (int it = 0; it < NIT; ++it) {
    const int buf = it & 1;
    if (it + 1 < NIT) gload((it + 1) * BK);     // global loads in flight

#pragma unroll
    for (int kk = 0; kk < 2; ++kk) {
      v16bf af[2], bf[4];
#pragma unroll
      for (int i = 0; i < 2; ++i) af[i] = load_A(&As[buf][wm * 32 + i * 16][kk * 32], LDT);
#pragma unroll
      for (int j = 0; j < 4; ++j) bf[j] = load_BT(&Ws[buf][wn * 64 + j * 16][kk * 32], LDT);
#pragma unroll
      for (int i = 0; i < 2; ++i)
#pragma unroll
        for (int j = 0; j < 4; ++j) acc[i][j] = wmma_bf16(af[i], bf[j], acc[i][j]);
    }

    if (it + 1 < NIT) lstore(buf ^ 1);          // other buffer: done being read
    __syncthreads();
  }

  // Epilogue. C layout: lane -> N=lane&15, VGPR r -> M = r + 8*(lane>>4).
  const int lane = tid & 31;
  const int nl   = lane & 15;
  const int mh   = (lane >> 4) * 8;
#pragma unroll
  for (int i = 0; i < 2; ++i)
#pragma unroll
    for (int j = 0; j < 4; ++j) {
      const int   gn  = n0 + wn * 64 + j * 16 + nl;
      const float bv  = bias[gn];
      const int   gm0 = m0 + wm * 32 + i * 16 + mh;   // 8-aligned
      if (MODE == 2) {
        const int b = gm0 >> 11, s = gm0 & (S_ - 1);
        const int h = gn >> 6,  d = gn & (DK_ - 1);
        v8bf pv;
#pragma unroll
        for (int r = 0; r < 8; ++r) pv[r] = (__bf16)(acc[i][j][r] + bv);
        *(v8bf*)&((__bf16*)Y)[((long)(b * H_ + h) * DK_ + d) * S_ + s] = pv;
      } else {
#pragma unroll
        for (int r = 0; r < 8; ++r) {
          const int   gm = gm0 + r;
          const float v  = acc[i][j][r] + bv;
          if (MODE == 0) {
            const int b = gm >> 11, s = gm & (S_ - 1);
            const int h = gn >> 6,  d = gn & (DK_ - 1);
            ((__bf16*)Y)[(((long)(b * H_ + h) * S_) + s) * DK_ + d] = (__bf16)v;
          } else {
            ((float*)Y)[(long)gm * D_ + gn] = v;
          }
        }
      }
    }
}

// ---------------------------------------------------------------------------
// Flash attention, transposed orientation:
//   S^T = K @ Q^T   (A = K rows,  B = Q rows via load_BT)
//   O^T = V^T @ P^T (A = Vt rows, B = P  rows via load_BT)
// Qh,Kh: [B,H,S,DK] bf16.  Vt: [B,H,DK,S] bf16.  AO: [B,S,D] bf16.
// K/V chunks double-buffered via async global->LDS loads (ASYNCcnt) when
// available; softmax stats are per-lane scalars + one shfl_xor(16).
// ---------------------------------------------------------------------------
__global__ __launch_bounds__(256)
void flash_attn(const __bf16* __restrict__ Qh, const __bf16* __restrict__ Kh,
                const __bf16* __restrict__ Vt, __bf16* __restrict__ AO) {
  constexpr int BM = 128, BT = 64, LDQ = DK_ + 8, LDV = BT + 8, LDP = BT + 8;
  constexpr int NCHUNK = S_ / BT;
  __shared__ __bf16 Qs[BM][LDQ];        // [query m][d]
  __shared__ __bf16 Ks[2][BT][LDQ];     // [buf][t][d]
  __shared__ __bf16 Vts[2][DK_][LDV];   // [buf][d][t]
  __shared__ __bf16 Ps[8][16][LDP];     // per wave: [query m][t]

  const int  tid   = threadIdx.x;
  const int  wave  = tid >> 5;
  const int  lane  = tid & 31;
  const int  bh    = blockIdx.y;                 // b*H + h
  const int  row0  = blockIdx.x * BM;
  const long base  = (long)bh * S_ * DK_;        // Qh/Kh
  const long vbase = (long)bh * DK_ * S_;        // Vt

  auto issue_chunk = [&](int t0, int bufidx) {
    const int r = tid >> 2, c = (tid & 3) * 16;  // 16 halves = 32B per thread
    const __bf16* ks = Kh + base  + (long)(t0 + r) * DK_ + c;
    const __bf16* vs = Vt + vbase + (long)r * S_ + t0 + c;
#if USE_ASYNC_LDS
    async_cp16B(ks,     &Ks[bufidx][r][c]);
    async_cp16B(ks + 8, &Ks[bufidx][r][c + 8]);
    async_cp16B(vs,     &Vts[bufidx][r][c]);
    async_cp16B(vs + 8, &Vts[bufidx][r][c + 8]);
#else
#pragma unroll
    for (int j = 0; j < 16; ++j) {
      Ks[bufidx][r][c + j]  = ks[j];
      Vts[bufidx][r][c + j] = vs[j];
    }
#endif
  };

  // Kick off the Q strip + first K/V chunk.
  {
    const int r = tid >> 1, c = (tid & 1) * 32;
    const __bf16* src = Qh + base + (long)(row0 + r) * DK_ + c;
#if USE_ASYNC_LDS
    async_cp16B(src,      &Qs[r][c]);
    async_cp16B(src + 8,  &Qs[r][c + 8]);
    async_cp16B(src + 16, &Qs[r][c + 16]);
    async_cp16B(src + 24, &Qs[r][c + 24]);
#else
#pragma unroll
    for (int j = 0; j < 32; ++j) Qs[r][c + j] = src[j];
#endif
  }
  issue_chunk(0, 0);
  wait_async0();
  __syncthreads();

  // Q as B-operand fragments (k = d).
  v16bf qb[2];
#pragma unroll
  for (int kk = 0; kk < 2; ++kk) qb[kk] = load_BT(&Qs[wave * 16][kk * 32], LDQ);

  v8f o[4];                                      // O^T tiles: rows d, cols m
#pragma unroll
  for (int j = 0; j < 4; ++j) o[j] = {};
  float mrun = -1e30f, lrun = 0.f;

  const float scale = 0.125f;                    // 1/sqrt(64)
  const int   mloc  = lane & 15;
  const int   hi8   = (lane >> 4) * 8;

  for (int it = 0; it < NCHUNK; ++it) {
    const int buf = it & 1;
    if (it + 1 < NCHUNK) issue_chunk((it + 1) * BT, buf ^ 1);

    // S^T tiles (rows t, cols m): st[ti] covers t = ti*16 + hi8 + r.
    v8f st[4];
#pragma unroll
    for (int ti = 0; ti < 4; ++ti) {
      v8f c = {};
#pragma unroll
      for (int kk = 0; kk < 2; ++kk)
        c = wmma_bf16(load_A(&Ks[buf][ti * 16][kk * 32], LDQ), qb[kk], c);
      st[ti] = c * scale;
    }

    // Online softmax: this lane's 32 values all belong to query `mloc`.
    float mx = st[0][0];
#pragma unroll
    for (int ti = 0; ti < 4; ++ti)
#pragma unroll
      for (int r = 0; r < 8; ++r) mx = fmaxf(mx, st[ti][r]);
    mx = fmaxf(mx, __shfl_xor(mx, 16, 32));
    const float mnew  = fmaxf(mrun, mx);
    const float alpha = __expf(mrun - mnew);
    mrun = mnew;
    float rs = 0.f;
#pragma unroll
    for (int ti = 0; ti < 4; ++ti)
#pragma unroll
      for (int r = 0; r < 8; ++r) {
        const float e = __expf(st[ti][r] - mnew);
        st[ti][r] = e;
        rs += e;
      }
    rs += __shfl_xor(rs, 16, 32);
    lrun = lrun * alpha + rs;
#pragma unroll
    for (int di = 0; di < 4; ++di)
#pragma unroll
      for (int r = 0; r < 8; ++r) o[di][r] *= alpha;

    // Stage P row-major [m][t]: 8 contiguous halves per ti -> ds_store_b128.
#pragma unroll
    for (int ti = 0; ti < 4; ++ti) {
      v8bf pv;
#pragma unroll
      for (int r = 0; r < 8; ++r) pv[r] = (__bf16)st[ti][r];
      *(v8bf*)&Ps[wave][mloc][ti * 16 + hi8] = pv;
    }

    // O^T += V^T @ P^T.
    v16bf pb[2];
#pragma unroll
    for (int kk = 0; kk < 2; ++kk) pb[kk] = load_BT(&Ps[wave][0][kk * 32], LDP);
#pragma unroll
    for (int di = 0; di < 4; ++di)
#pragma unroll
      for (int kk = 0; kk < 2; ++kk)
        o[di] = wmma_bf16(load_A(&Vts[buf][di * 16][kk * 32], LDV), pb[kk], o[di]);

    wait_async0();       // next chunk landed (this wave's async ops)
    __syncthreads();     // all waves done reading current buffers
  }

  // Epilogue: O^T rows d are contiguous per lane -> 16B bf16 stores.
  {
    const int   b    = bh >> 4, h = bh & 15;
    const float linv = 1.f / lrun;
    const long  row  = ((long)b * S_ + row0 + wave * 16 + mloc) * D_ + h * DK_;
#pragma unroll
    for (int di = 0; di < 4; ++di) {
      v8bf pv;
#pragma unroll
      for (int r = 0; r < 8; ++r) pv[r] = (__bf16)(o[di][r] * linv);
      *(v8bf*)&AO[row + di * 16 + hi8] = pv;
    }
  }
}

// ---------------------------------------------------------------------------
// Launch
// ---------------------------------------------------------------------------
extern "C" void kernel_launch(void* const* d_in, const int* in_sizes, int n_in,
                              void* d_out, int out_size, void* d_ws, size_t ws_size,
                              hipStream_t stream) {
  const float* q  = (const float*)d_in[0];
  const float* k  = (const float*)d_in[1];
  const float* v  = (const float*)d_in[2];
  const float* wq = (const float*)d_in[3];
  const float* bq = (const float*)d_in[4];
  const float* wk = (const float*)d_in[5];
  const float* bk = (const float*)d_in[6];
  const float* wv = (const float*)d_in[7];
  const float* bv = (const float*)d_in[8];
  const float* wo = (const float*)d_in[9];
  const float* bo = (const float*)d_in[10];

  const size_t elems = (size_t)MTOT * D_;
  __bf16* Qh  = (__bf16*)d_ws;
  __bf16* Kh  = Qh + elems;
  __bf16* Vtr = Kh + elems;      // [B,H,DK,S]
  __bf16* AO  = Vtr + elems;

  dim3 blk(256);
  dim3 gGemm(D_ / 128, MTOT / 128);              // (8, 64)
  dim3 gAttn(S_ / 128, B_ * H_);                 // (16, 64)

  gemm_xwt<float, 0><<<gGemm, blk, 0, stream>>>(q, wq, bq, (void*)Qh);
  gemm_xwt<float, 0><<<gGemm, blk, 0, stream>>>(k, wk, bk, (void*)Kh);
  gemm_xwt<float, 2><<<gGemm, blk, 0, stream>>>(v, wv, bv, (void*)Vtr);
  flash_attn<<<gAttn, blk, 0, stream>>>(Qh, Kh, Vtr, AO);
  gemm_xwt<__bf16, 1><<<gGemm, blk, 0, stream>>>(AO, wo, bo, d_out);
}
